// CausalSelfAttention_52630529245535
// MI455X (gfx1250) — matrix operand order repaired and checked by reading, not compiled, not run on hardware
//
#include <hip/hip_runtime.h>
#include <stdint.h>

// ---------------------------------------------------------------------------
// Causal self-attention for MI455X (gfx1250): f16 WMMA pipeline.
// x[B,T,C] @ Wqkv -> q,k,v -> flash attention (causal, online softmax) -> proj
// B=2, T=2048, C=1024, H=16, D=64
// GEMMs stage shared B tiles through LDS with GLOBAL_LOAD_ASYNC_TO_LDS_B128.
// ---------------------------------------------------------------------------

#define DEVFN __device__ __forceinline__

typedef _Float16 v16h __attribute__((ext_vector_type(16)));
typedef _Float16 h8   __attribute__((ext_vector_type(8)));
typedef float    v8f  __attribute__((ext_vector_type(8)));

constexpr int  Bb = 2, Tt = 2048, Cc = 1024, Hh = 16, Dd = 64;
constexpr int  N_QKV = 3 * Cc;            // 3072
constexpr int  Mrows = Bb * Tt;           // 4096
constexpr float ATTN_SCALE = 0.125f;      // D^-0.5

DEVFN v8f wmma16(v16h a, v16h b, v8f c) {
  // v_wmma_f32_16x16x32_f16: (neg_a, A, neg_b, B, c_mod, C, reuse_a, reuse_b)
  return __builtin_amdgcn_wmma_f32_16x16x32_f16(false, a, false, b, (short)0, c,
                                                false, false);
}

DEVFN v16h combine8(h8 lo, h8 hi) {
  v16h r;
#pragma unroll
  for (int i = 0; i < 8; ++i) { r[i] = lo[i]; r[i + 8] = hi[i]; }
  return r;
}

// A-fragment (16x32 f16, M x K). ISA layout: low lane group holds K 0..7 then
// 16..23, high group K 8..15 then 24..31 -> two contiguous 16B loads per lane.
DEVFN v16h load_afrag(const _Float16* __restrict__ base, int stride, int row0,
                      int k0) {
  int lane = threadIdx.x & 31, m = lane & 15, g = lane >> 4;
  const _Float16* p = base + (size_t)(row0 + m) * stride + k0;
  h8 lo = *(const h8*)(p + g * 8);
  h8 hi = *(const h8*)(p + 16 + g * 8);
  return combine8(lo, hi);
}

// B-fragment (32x16 f16, K x N), source stored column-per-row (baseT[n][k]):
// lane n holds column n; low lane group K 0..7 then 16..23, high group K 8..15 then 24..31.
DEVFN v16h load_bfrag(const _Float16* __restrict__ baseT, int stride, int col0,
                      int k0) {
  int lane = threadIdx.x & 31, n = lane & 15, g = lane >> 4;
  const _Float16* p = baseT + (size_t)(col0 + n) * stride + k0;
  h8 lo = *(const h8*)(p + g * 8);
  h8 hi = *(const h8*)(p + 16 + g * 8);
  return combine8(lo, hi);
}

// B-fragment sourced from an LDS tile ldsB[64][32] (row = col0+n, halfs 0..31).
DEVFN v16h load_bfrag_lds(const _Float16* ldsB, int col0) {
  int lane = threadIdx.x & 31, n = lane & 15, g = lane >> 4;
  const _Float16* p = ldsB + (col0 + n) * 32;
  h8 lo = *(const h8*)(p + g * 8);
  h8 hi = *(const h8*)(p + 16 + g * 8);
  return combine8(lo, hi);
}

// Per-lane async copy of 16 bytes global -> LDS (tracked by ASYNCcnt).
DEVFN void async_copy16(const _Float16* src, _Float16* ldsDst) {
  unsigned lds_off = (unsigned)(uintptr_t)ldsDst;          // low 32b = LDS addr
  unsigned long long ga = (unsigned long long)(uintptr_t)src;
  asm volatile("global_load_async_to_lds_b128 %0, %1, off"
               :: "v"(lds_off), "v"(ga) : "memory");
}
DEVFN void wait_async0() {
  asm volatile("s_wait_asynccnt 0" ::: "memory");
}

// ---------------------------------------------------------------------------
// 1) conversion / transpose kernels
// ---------------------------------------------------------------------------
__global__ void cvt_f16_kernel(const float* __restrict__ in,
                               _Float16* __restrict__ out, int n) {
  for (int i = blockIdx.x * blockDim.x + threadIdx.x; i < n;
       i += gridDim.x * blockDim.x)
    out[i] = (_Float16)in[i];
}

// in[K][N] row-major -> outT[N][K]  (coalesced writes over k)
__global__ void transpose_cvt_kernel(const float* __restrict__ in,
                                     _Float16* __restrict__ outT, int K, int N) {
  int total = K * N;
  for (int i = blockIdx.x * blockDim.x + threadIdx.x; i < total;
       i += gridDim.x * blockDim.x) {
    int n = i / K, k = i - n * K;
    outT[i] = (_Float16)in[(size_t)k * N + n];
  }
}

// ---------------------------------------------------------------------------
// 2) QKV GEMM: [4096,1024] x [1024,3072] + bias, scatter into q,k (BHTD) and
//    v transposed (BHDT), f16.  8 waves/block share one 64-col B tile staged
//    through LDS by async copies.
// ---------------------------------------------------------------------------
__global__ __launch_bounds__(256) void gemm_qkv_kernel(
    const _Float16* __restrict__ xh, const _Float16* __restrict__ wT,
    const float* __restrict__ bias, _Float16* __restrict__ qh,
    _Float16* __restrict__ kh, _Float16* __restrict__ vt) {
  __shared__ __align__(16) _Float16 ldsB[64 * 32];   // 4 KB B tile
  int wid = threadIdx.x >> 5;
  int miBlk = blockIdx.x & 31;        // 256/8 row-tile groups
  int ni = blockIdx.x >> 5;           // 3072/64 = 48 col tiles
  int mi = miBlk * 8 + wid;
  int lane = threadIdx.x & 31, hi = lane >> 4, nl = lane & 15;
  int cRow = threadIdx.x >> 2;        // 0..63   : staging assignment
  int cCol = (threadIdx.x & 3) * 8;   // 0,8,16,24 halfs
  const _Float16* srcB = wT + (size_t)(ni * 64 + cRow) * Cc + cCol;
  _Float16* dstB = &ldsB[cRow * 32 + cCol];

  v8f acc0 = {}, acc1 = {}, acc2 = {}, acc3 = {};
  for (int k = 0; k < Cc; k += 32) {
    __syncthreads();                          // previous tile consumed
    async_copy16(srcB + k, dstB);             // 256 lanes x 16B = 4 KB
    __builtin_prefetch(srcB + k + 64, 0, 0);  // global_prefetch_b8, next line
    wait_async0();
    __syncthreads();                          // tile visible to all waves

    v16h a = load_afrag(xh, Cc, mi * 16, k);
    acc0 = wmma16(a, load_bfrag_lds(ldsB,  0), acc0);
    acc1 = wmma16(a, load_bfrag_lds(ldsB, 16), acc1);
    acc2 = wmma16(a, load_bfrag_lds(ldsB, 32), acc2);
    acc3 = wmma16(a, load_bfrag_lds(ldsB, 48), acc3);
  }
#pragma unroll
  for (int nt = 0; nt < 4; ++nt) {
    v8f acc = (nt == 0) ? acc0 : (nt == 1) ? acc1 : (nt == 2) ? acc2 : acc3;
    int n_g = ni * 64 + nt * 16 + nl;
    int which = n_g >> 10;            // 0=q 1=k 2=v
    int cc = n_g & 1023;
    int hhd = cc >> 6, d = cc & 63;
#pragma unroll
    for (int r = 0; r < 8; ++r) {
      int m_g = mi * 16 + r + 8 * hi;
      int b = m_g >> 11, t = m_g & 2047;
      _Float16 v = (_Float16)(acc[r] + bias[n_g]);
      size_t bh = (size_t)(b * Hh + hhd);
      if (which == 0)      qh[(bh * Tt + t) * Dd + d] = v;
      else if (which == 1) kh[(bh * Tt + t) * Dd + d] = v;
      else                 vt[(bh * Dd + d) * Tt + t] = v;
    }
  }
}

// ---------------------------------------------------------------------------
// 3) Flash attention: one wave per 16 query rows, key chunks of 32.
// ---------------------------------------------------------------------------
__global__ __launch_bounds__(256) void attn_kernel(
    const _Float16* __restrict__ qh, const _Float16* __restrict__ kh,
    const _Float16* __restrict__ vt, _Float16* __restrict__ oh) {
  __shared__ __align__(16) _Float16 plds[8][16 * 32];   // per-wave P tile
  int wid = threadIdx.x >> 5;
  int lane = threadIdx.x & 31, hi = lane >> 4, nl = lane & 15;
  int bh = blockIdx.x >> 4;             // B*H = 32
  int qblk = blockIdx.x & 15;           // T/128 = 16
  int q0 = qblk * 128 + wid * 16;

  const _Float16* Q = qh + (size_t)bh * Tt * Dd;
  const _Float16* K = kh + (size_t)bh * Tt * Dd;
  const _Float16* V = vt + (size_t)bh * Dd * Tt;

  v16h qa0 = load_afrag(Q, Dd, q0, 0);
  v16h qa1 = load_afrag(Q, Dd, q0, 32);

  v8f o0 = {}, o1 = {}, o2 = {}, o3 = {};
  float mrow[8], lrow[8];
#pragma unroll
  for (int r = 0; r < 8; ++r) { mrow[r] = -1e30f; lrow[r] = 0.f; }

  int kend = q0 + 16;                   // last key = last query row (causal)
  for (int kb = 0; kb < kend; kb += 32) {
    // S = Q K^T  (two 16x16 tiles covering 32 keys; K-dim = D = 64)
    v8f s0 = {}, s1 = {};
    s0 = wmma16(qa0, load_bfrag(K, Dd, kb, 0), s0);
    s0 = wmma16(qa1, load_bfrag(K, Dd, kb, 32), s0);
    s1 = wmma16(qa0, load_bfrag(K, Dd, kb + 16, 0), s1);
    s1 = wmma16(qa1, load_bfrag(K, Dd, kb + 16, 32), s1);

    // scale + causal mask (data-only, keeps EXEC all-1s)
    float nm[8];
#pragma unroll
    for (int r = 0; r < 8; ++r) {
      int qrow = q0 + r + 8 * hi;
      float a0 = (kb + nl <= qrow) ? s0[r] * ATTN_SCALE : -1e30f;
      float a1 = (kb + 16 + nl <= qrow) ? s1[r] * ATTN_SCALE : -1e30f;
      s0[r] = a0; s1[r] = a1;
      float t = fmaxf(a0, a1);
      t = fmaxf(t, __shfl_xor(t, 1));
      t = fmaxf(t, __shfl_xor(t, 2));
      t = fmaxf(t, __shfl_xor(t, 4));
      t = fmaxf(t, __shfl_xor(t, 8));
      nm[r] = fmaxf(mrow[r], t);
    }
    // exp, rescale running state, stage P tile to LDS (D-layout -> row major)
#pragma unroll
    for (int r = 0; r < 8; ++r) {
      float p0 = __expf(s0[r] - nm[r]);
      float p1 = __expf(s1[r] - nm[r]);
      float alpha = __expf(mrow[r] - nm[r]);
      mrow[r] = nm[r];
      float rs = p0 + p1;
      rs += __shfl_xor(rs, 1);
      rs += __shfl_xor(rs, 2);
      rs += __shfl_xor(rs, 4);
      rs += __shfl_xor(rs, 8);
      lrow[r] = lrow[r] * alpha + rs;
      o0[r] *= alpha; o1[r] *= alpha; o2[r] *= alpha; o3[r] *= alpha;
      int row = r + 8 * hi;
      plds[wid][row * 32 + nl]      = (_Float16)p0;
      plds[wid][row * 32 + 16 + nl] = (_Float16)p1;
    }
    // within-wave LDS RAW: DS ops are in-order; fence the compiler + hw
    asm volatile("s_wait_dscnt 0" ::: "memory");

    // reload P as an A-fragment (16x32) from LDS
    const _Float16* pb = &plds[wid][0];
    h8 plo = *(const h8*)(pb + nl * 32 + hi * 8);
    h8 phi = *(const h8*)(pb + nl * 32 + 16 + hi * 8);
    v16h pf = combine8(plo, phi);

    // O += P V   (V stored transposed [D][T] -> contiguous B-fragments)
    o0 = wmma16(pf, load_bfrag(V, Tt, 0, kb), o0);
    o1 = wmma16(pf, load_bfrag(V, Tt, 16, kb), o1);
    o2 = wmma16(pf, load_bfrag(V, Tt, 32, kb), o2);
    o3 = wmma16(pf, load_bfrag(V, Tt, 48, kb), o3);
  }

  // normalize + store to [B,T,C] f16 (col = h*64 + nt*16 + nl)
  int b = bh >> 4, hhd = bh & 15;
#pragma unroll
  for (int r = 0; r < 8; ++r) {
    float inv = 1.0f / lrow[r];
    int t = q0 + r + 8 * hi;
    size_t row = ((size_t)b * Tt + t) * Cc + hhd * 64 + nl;
    oh[row +  0] = (_Float16)(o0[r] * inv);
    oh[row + 16] = (_Float16)(o1[r] * inv);
    oh[row + 32] = (_Float16)(o2[r] * inv);
    oh[row + 48] = (_Float16)(o3[r] * inv);
  }
}

// ---------------------------------------------------------------------------
// 4) output projection: [4096,1024] x [1024,1024] + bias -> f32 out
//    (same async-staged LDS B tile as the QKV GEMM)
// ---------------------------------------------------------------------------
__global__ __launch_bounds__(256) void gemm_proj_kernel(
    const _Float16* __restrict__ ah, const _Float16* __restrict__ wT,
    const float* __restrict__ bias, float* __restrict__ out) {
  __shared__ __align__(16) _Float16 ldsB[64 * 32];   // 4 KB B tile
  int wid = threadIdx.x >> 5;
  int miBlk = blockIdx.x & 31;        // 32 row-tile groups
  int ni = blockIdx.x >> 5;           // 1024/64 = 16 col tiles
  int mi = miBlk * 8 + wid;
  int lane = threadIdx.x & 31, hi = lane >> 4, nl = lane & 15;
  int cRow = threadIdx.x >> 2;
  int cCol = (threadIdx.x & 3) * 8;
  const _Float16* srcB = wT + (size_t)(ni * 64 + cRow) * Cc + cCol;
  _Float16* dstB = &ldsB[cRow * 32 + cCol];

  v8f acc0 = {}, acc1 = {}, acc2 = {}, acc3 = {};
  for (int k = 0; k < Cc; k += 32) {
    __syncthreads();
    async_copy16(srcB + k, dstB);
    __builtin_prefetch(srcB + k + 64, 0, 0);
    wait_async0();
    __syncthreads();

    v16h a = load_afrag(ah, Cc, mi * 16, k);
    acc0 = wmma16(a, load_bfrag_lds(ldsB,  0), acc0);
    acc1 = wmma16(a, load_bfrag_lds(ldsB, 16), acc1);
    acc2 = wmma16(a, load_bfrag_lds(ldsB, 32), acc2);
    acc3 = wmma16(a, load_bfrag_lds(ldsB, 48), acc3);
  }
#pragma unroll
  for (int nt = 0; nt < 4; ++nt) {
    v8f acc = (nt == 0) ? acc0 : (nt == 1) ? acc1 : (nt == 2) ? acc2 : acc3;
    int n_g = ni * 64 + nt * 16 + nl;
    float bn = bias[n_g];
#pragma unroll
    for (int r = 0; r < 8; ++r) {
      int m_g = mi * 16 + r + 8 * hi;
      out[(size_t)m_g * Cc + n_g] = acc[r] + bn;
    }
  }
}

// ---------------------------------------------------------------------------
extern "C" void kernel_launch(void* const* d_in, const int* in_sizes, int n_in,
                              void* d_out, int out_size, void* d_ws,
                              size_t ws_size, hipStream_t stream) {
  const float* x      = (const float*)d_in[0];  // [B,T,C]
  const float* w_qkv  = (const float*)d_in[1];  // [C,3C]
  const float* b_qkv  = (const float*)d_in[2];  // [3C]
  const float* w_proj = (const float*)d_in[3];  // [C,C]
  const float* b_proj = (const float*)d_in[4];  // [C]
  float* out = (float*)d_out;

  // workspace layout (f16 elements)
  _Float16* xh    = (_Float16*)d_ws;                     // 4,194,304
  _Float16* wqkvT = xh + (size_t)Mrows * Cc;             // 3,145,728
  _Float16* wprjT = wqkvT + (size_t)N_QKV * Cc;          // 1,048,576
  _Float16* qh    = wprjT + (size_t)Cc * Cc;             // 4,194,304
  _Float16* kh    = qh + (size_t)Bb * Hh * Tt * Dd;
  _Float16* vt    = kh + (size_t)Bb * Hh * Tt * Dd;
  _Float16* ohalf = vt + (size_t)Bb * Hh * Tt * Dd;      // total ~50.3 MB

  // 1) convert / transpose weights and activations to f16
  cvt_f16_kernel<<<2048, 256, 0, stream>>>(x, xh, Mrows * Cc);
  transpose_cvt_kernel<<<2048, 256, 0, stream>>>(w_qkv, wqkvT, Cc, N_QKV);
  transpose_cvt_kernel<<<1024, 256, 0, stream>>>(w_proj, wprjT, Cc, Cc);

  // 2) qkv = x @ Wqkv + b   (32 row groups x 48 col tiles)
  gemm_qkv_kernel<<<1536, 256, 0, stream>>>(xh, wqkvT, b_qkv, qh, kh, vt);

  // 3) flash attention: B*H * (T/128) blocks, 8 waves each
  attn_kernel<<<512, 256, 0, stream>>>(qh, kh, vt, ohalf);

  // 4) out = attn @ Wproj + b   (32 row groups x 16 col tiles)
  gemm_proj_kernel<<<512, 256, 0, stream>>>(ohalf, wprjT, b_proj, out);
}
